// RoofN3DNet_56109452755397
// MI455X (gfx1250) — matrix operand order repaired and checked by reading, not compiled, not run on hardware
//
#include <hip/hip_runtime.h>
#include <hip/hip_bf16.h>

// ======================================================================
// RoofN3D-style point network forward for MI455X (gfx1250, wave32).
//
// Every GEMM runs through one branch-free WMMA kernel
// (v_wmma_f32_16x16x32_bf16, f32 accumulate):
//   - A operands are bf16, row stride = K rounded up to 32, pads zeroed
//     by their producers.
//   - B weights are pre-packed into k_gemm's B lane layout (contiguous k)
//     (zero-padded), so the B fetch is one 32B vector load per lane.
//   - C stores are unconditional into padded buffers; pad columns come
//     out exactly zero because B/bias pads are zero.
// No lane-dependent control flow anywhere in the hot loop.
//
// Working set (~156 MB incl. the 100 MB bf16 f-matrix) fits the 192 MB
// global L2, so inter-kernel traffic stays on-chip; at 23.3 TB/s HBM the
// network (~10 GFLOP, GEMM-shaped) is WMMA-pipe work, not bandwidth work.
//
// jax.random permutation / uniform draws are replaced by deterministic
// substitutes (bijective affine index maps, stratified uniforms).
// ======================================================================

typedef __attribute__((ext_vector_type(16))) __bf16 v16bf;
typedef __attribute__((ext_vector_type(8)))  __bf16 v8bf;
typedef __attribute__((ext_vector_type(8)))  float  v8f;

#define N_PTS   16384
#define C_PL2   96      // 2*PL
#define NC      16
#define RF_DIM  99      // 2*PL + 3
#define RF_LD   128     // RF_DIM rounded to 32

// ----------------------------------------------------------------------
// Deterministic point-subset selection (substitute for jax permutations).
// ----------------------------------------------------------------------
__global__ void k_prep_sel(const float* __restrict__ pts,
                           int* __restrict__ sel1, int* __restrict__ sel2, int* __restrict__ sel3,
                           float* __restrict__ p1, float* __restrict__ p2, float* __restrict__ p3)
{
    int i = threadIdx.x;
    if (i < 64) {
        int s = (i * 2531 + 7) & (N_PTS - 1);   // odd multiplier -> bijective mod 16384
        sel1[i] = s;
        p1[i*3+0] = pts[s*3+0]; p1[i*3+1] = pts[s*3+1]; p1[i*3+2] = pts[s*3+2];
    }
    __syncthreads();
    if (i < 16) {
        int s = (i * 13 + 5) & 63;
        sel2[i] = s;
        p2[i*3+0] = p1[s*3+0]; p2[i*3+1] = p1[s*3+1]; p2[i*3+2] = p1[s*3+2];
    }
    __syncthreads();
    if (i < 8) {
        int s = (i * 5 + 3) & 15;
        sel3[i] = s;
        p3[i*3+0] = p2[s*3+0]; p3[i*3+1] = p2[s*3+1]; p3[i*3+2] = p2[s*3+2];
    }
}

// ----------------------------------------------------------------------
// Brute-force KNN: one thread per query, insertion-sorted top-K (K<=8).
// ----------------------------------------------------------------------
__global__ void k_knn(const float* __restrict__ Q, int M,
                      const float* __restrict__ P, int Np, int K,
                      int* __restrict__ idx)
{
    int m = blockIdx.x * blockDim.x + threadIdx.x;
    if (m >= M) return;
    float qx = Q[3*m], qy = Q[3*m+1], qz = Q[3*m+2];
    float bd[8]; int bi[8];
    #pragma unroll
    for (int i = 0; i < 8; ++i) { bd[i] = 3.4e38f; bi[i] = 0; }
    for (int p = 0; p < Np; ++p) {
        float dx = P[3*p]   - qx;
        float dy = P[3*p+1] - qy;
        float dz = P[3*p+2] - qz;
        float d  = dx*dx + dy*dy + dz*dz;
        if (d < bd[K-1]) {
            int j = K - 1;
            while (j > 0 && bd[j-1] > d) { bd[j] = bd[j-1]; bi[j] = bi[j-1]; --j; }
            bd[j] = d; bi[j] = p;
        }
    }
    for (int i = 0; i < K; ++i) idx[m*K + i] = bi[i];
}

// ----------------------------------------------------------------------
// rel / maxi normalization + center differences -> Dp (M*K x 64 bf16,
// cols 48..63 zero).
// ----------------------------------------------------------------------
__global__ void k_dists(const float* __restrict__ Q, int M,
                        const float* __restrict__ P,
                        const int* __restrict__ idx, int K,
                        const float* __restrict__ centers,
                        __hip_bfloat16* __restrict__ D)
{
    int m = blockIdx.x * blockDim.x + threadIdx.x;
    if (m >= M) return;
    float qx = Q[3*m], qy = Q[3*m+1], qz = Q[3*m+2];
    float rx[8], ry[8], rz[8];
    float mx = 0.f;
    for (int k = 0; k < K; ++k) {
        int p = idx[m*K + k];
        rx[k] = P[3*p]   - qx;
        ry[k] = P[3*p+1] - qy;
        rz[k] = P[3*p+2] - qz;
        float s = rx[k]*rx[k] + ry[k]*ry[k] + rz[k]*rz[k];
        mx = fmaxf(mx, s);
    }
    float maxi = sqrtf(mx);
    if (maxi == 0.f) maxi = 1.f;
    float inv = 1.f / maxi;
    for (int k = 0; k < K; ++k) {
        size_t base = (size_t)(m*K + k) * 64;
        float r0 = rx[k]*inv, r1 = ry[k]*inv, r2 = rz[k]*inv;
        #pragma unroll
        for (int n = 0; n < 16; ++n) {
            D[base +      n] = __float2bfloat16(r0 - centers[     n]);
            D[base + 16 + n] = __float2bfloat16(r1 - centers[16 + n]);
            D[base + 32 + n] = __float2bfloat16(r2 - centers[32 + n]);
            D[base + 48 + n] = __float2bfloat16(0.f);
        }
    }
}

// ----------------------------------------------------------------------
// Pack B (f32, NK or KN layout) into k_gemm's B lane layout (contiguous k), bf16, zero-padded:
//   Bp[((nt*Kc + kc)<<9) + (lane<<4) + e] = B(n = nt*16 + (lane&15),
//                                             k = kc*32 + (lane>>4)*16 + e)
// ----------------------------------------------------------------------
__global__ void k_pack_b(const float* __restrict__ B, int b_nk, int ldb,
                         int N, int K, int Kc, int total,
                         __hip_bfloat16* __restrict__ Bp)
{
    int t = blockIdx.x * blockDim.x + threadIdx.x;
    if (t >= total) return;
    int e    = t & 15;
    int lane = (t >> 4) & 31;
    int rest = t >> 9;
    int kc   = rest % Kc;
    int nt   = rest / Kc;
    int n = (nt << 4) + (lane & 15);
    int k = kc * 32 + ((lane >> 4) << 4) + e;
    float v = 0.f;
    if (n < N && k < K)
        v = b_nk ? B[(size_t)n*ldb + k] : B[(size_t)k*ldb + n];
    Bp[t] = __float2bfloat16(v);
}

__global__ void k_pack_bias(const float* __restrict__ b, int N, int Np,
                            float* __restrict__ bp)
{
    int t = threadIdx.x;
    if (t < Np) bp[t] = (b != nullptr && t < N) ? b[t] : 0.f;
}

// ----------------------------------------------------------------------
// Branch-free WMMA GEMM on padded operands.
//   A: bf16, lda = 32*Kc, rows >= 16*Mt readable.
//   Bp: packed by k_pack_b.  biasP: Np f32 (zero-padded).
//   C: padded, ldc >= 16*Nt; stores unconditional.
// ----------------------------------------------------------------------
template<int CBF16>
__global__ void k_gemm(const __hip_bfloat16* __restrict__ A, int lda,
                       int Mt, int Nt, int Kc,
                       const __hip_bfloat16* __restrict__ Bp,
                       const float* __restrict__ biasP,
                       float scale, int relu,
                       void* __restrict__ Cv, int ldc)
{
    int wave = (int)((blockIdx.x * blockDim.x + threadIdx.x) >> 5);
    if (wave >= Mt * Nt) return;            // wave-uniform: EXEC stays full
    int mt = wave / Nt, nt = wave % Nt;

    int lane = threadIdx.x & 31;
    int row  = lane & 15;
    int half = lane >> 4;

    const __hip_bfloat16* Arow = A  + (size_t)((mt << 4) + row) * lda;
    const __hip_bfloat16* Bt   = Bp + ((size_t)nt * Kc << 9) + (lane << 4);

    v8f acc = {0.f,0.f,0.f,0.f,0.f,0.f,0.f,0.f};

    for (int kc = 0; kc < Kc; ++kc) {
        int k0 = kc << 5;
        v8bf  lo = *(const v8bf *)(const void*)(Arow + k0 +     (half << 4));
        v8bf  hi = *(const v8bf *)(const void*)(Arow + k0 + 8 + (half << 4));
        v16bf b  = *(const v16bf*)(const void*)(Bt + ((size_t)kc << 9));
        v16bf a  = __builtin_shufflevector(lo, hi, 0,1,2,3,4,5,6,7,
                                                   8,9,10,11,12,13,14,15);
        __builtin_prefetch((const char*)(Arow + k0 + 128), 0, 1);   // speculative
        acc = __builtin_amdgcn_wmma_f32_16x16x32_bf16(false, a, false, b,
                                                      (short)0, acc, false, false);
    }

    // D layout: VGPR r, lanes 0-15 -> M=r, lanes 16-31 -> M=8+r; N = lane&15
    int n = (nt << 4) + row;
    float bias = biasP[n];
    #pragma unroll
    for (int r = 0; r < 8; ++r) {
        int m = (mt << 4) + r + (half << 3);
        float v = acc[r] * scale + bias;
        if (relu) v = fmaxf(v, 0.f);
        if (CBF16)
            ((__hip_bfloat16*)Cv)[(size_t)m*ldc + n] = __float2bfloat16(v);
        else
            ((float*)Cv)[(size_t)m*ldc + n] = v;
    }
}

// ----------------------------------------------------------------------
// f[m, c*16+n] = sum_k feat[idx[m,k], c] * H[(m*K+k), n]  -> bf16, ld ldF.
// ----------------------------------------------------------------------
__global__ void k_feinsum(const float* __restrict__ feat, int Cin,
                          const int* __restrict__ idx, int K,
                          const __hip_bfloat16* __restrict__ H, int M,
                          __hip_bfloat16* __restrict__ F, int ldF)
{
    long t = (long)blockIdx.x * blockDim.x + threadIdx.x;
    if (t >= (long)M * Cin) return;
    int m = (int)(t / Cin), c = (int)(t % Cin);
    float acc[16];
    #pragma unroll
    for (int n = 0; n < 16; ++n) acc[n] = 0.f;
    for (int k = 0; k < K; ++k) {
        float g = feat[(size_t)idx[m*K + k] * Cin + c];
        const __hip_bfloat16* h = &H[(size_t)(m*K + k) * 16];
        #pragma unroll
        for (int n = 0; n < 16; ++n) acc[n] += g * __bfloat162float(h[n]);
    }
    size_t base = (size_t)m * ldF + (size_t)c * 16;
    #pragma unroll
    for (int n = 0; n < 16; ++n) F[base + n] = __float2bfloat16(acc[n]);
}

__global__ void k_zero_bf(__hip_bfloat16* __restrict__ X, int M, int ld, int c0)
{
    int span = ld - c0;
    int t = blockIdx.x * blockDim.x + threadIdx.x;
    if (t >= M * span) return;
    int m = t / span, c = c0 + t % span;
    X[(size_t)m*ld + c] = __float2bfloat16(0.f);
}

// ----------------------------------------------------------------------
// BatchNorm over M rows (per-channel) + ReLU; optional bf16 mirror output.
// ----------------------------------------------------------------------
__global__ void k_bn_relu(const float* __restrict__ X, int M, int C,
                          const float* __restrict__ g, const float* __restrict__ b,
                          float* __restrict__ Y, int ldy,
                          __hip_bfloat16* __restrict__ Ybf)
{
    __shared__ float s1[256], s2[256];
    int c = blockIdx.x;
    int tid = threadIdx.x;
    float s = 0.f, ss = 0.f;
    for (int m = tid; m < M; m += blockDim.x) {
        float v = X[(size_t)m*C + c];
        s += v; ss += v*v;
    }
    s1[tid] = s; s2[tid] = ss;
    __syncthreads();
    for (int st = blockDim.x >> 1; st > 0; st >>= 1) {
        if (tid < st) { s1[tid] += s1[tid+st]; s2[tid] += s2[tid+st]; }
        __syncthreads();
    }
    float mean = s1[0] / M;
    float var  = s2[0] / M - mean*mean;
    float inv  = rsqrtf(var + 1e-5f);
    float gg = g[c], bb = b[c];
    for (int m = tid; m < M; m += blockDim.x) {
        float v = fmaxf((X[(size_t)m*C + c] - mean) * inv * gg + bb, 0.f);
        Y[(size_t)m*ldy + c] = v;
        if (Ybf) Ybf[(size_t)m*ldy + c] = __float2bfloat16(v);
    }
}

__global__ void k_copy(const float* __restrict__ src, int M, int C,
                       float* __restrict__ dst, int ldd)
{
    int t = blockIdx.x * blockDim.x + threadIdx.x;
    if (t >= M*C) return;
    int m = t / C, c = t % C;
    dst[(size_t)m*ldd + c] = src[(size_t)m*C + c];
}

// Extract valid (M x C) region from padded f32 (lds) into dst (ldd).
__global__ void k_extract(const float* __restrict__ src, int lds, int M, int C,
                          float* __restrict__ dst, int ldd)
{
    int t = blockIdx.x * blockDim.x + threadIdx.x;
    if (t >= M*C) return;
    int m = t / C, c = t % C;
    dst[(size_t)m*ldd + c] = src[(size_t)m*lds + c];
}

__global__ void k_argmax(const float* __restrict__ xout, int lds, int M, int C,
                         int* __restrict__ cls)
{
    int m = blockIdx.x * blockDim.x + threadIdx.x;
    if (m >= M) return;
    float best = xout[(size_t)m*lds]; int bi = 0;
    for (int c = 1; c < C; ++c) {
        float v = xout[(size_t)m*lds + c];
        if (v > best) { best = v; bi = c; }   // first max wins (jnp.argmax)
    }
    cls[m] = bi;
}

// ----------------------------------------------------------------------
// Per-class sampling of 32 points; gathers rf = [x1d(96)|pts(3)] as bf16
// padded to 128 cols (zeros beyond 99).
// ----------------------------------------------------------------------
__global__ void k_sample(const int* __restrict__ cls, int N,
                         const float* __restrict__ x1d,
                         const float* __restrict__ pts,
                         __hip_bfloat16* __restrict__ rf)
{
    __shared__ int sSel[32];
    __shared__ int sCount;
    int c = blockIdx.x + 1;
    if (threadIdx.x == 0) {
        int count = 0;
        for (int n = 0; n < N; ++n) count += (cls[n] == c);
        sCount = count;
        int j[32];
        for (int t = 0; t < 32; ++t) {
            float u = (2.f*t + 1.f) / 64.f;      // deterministic uniform substitute
            int jj = (int)floorf(u * (float)count);
            if (jj > count - 1) jj = count - 1;
            if (jj < 0) jj = 0;
            j[t] = jj;
        }
        int r = 0, t = 0;
        for (int n = 0; n < N && t < 32; ++n) {
            if (cls[n] == c) {
                while (t < 32 && j[t] == r) { sSel[t] = n; ++t; }
                ++r;
            }
        }
        for (; t < 32; ++t) sSel[t] = 0;
    }
    __syncthreads();
    int count = sCount;
    for (int q = threadIdx.x; q < 32*RF_LD; q += blockDim.x) {
        int row = q >> 7, col = q & (RF_LD - 1);
        float v = 0.f;
        if (count > 0 && col < RF_DIM) {
            int p = sSel[row];
            v = (col < C_PL2) ? x1d[(size_t)p*C_PL2 + col]
                              : pts[(size_t)p*3 + (col - C_PL2)];
        }
        rf[(size_t)blockIdx.x * 32 * RF_LD + q] = __float2bfloat16(v);
    }
}

__global__ void k_meanrows(const float* __restrict__ X, int M, int C,
                           __hip_bfloat16* __restrict__ out)
{
    int c = blockIdx.x * blockDim.x + threadIdx.x;
    if (c >= C) return;
    float s = 0.f;
    for (int m = 0; m < M; ++m) s += X[(size_t)m*C + c];
    out[c] = __float2bfloat16(s / M);
}

// ======================================================================
// Host orchestration
// ======================================================================
struct CvP { const float *w,*cen,*l1W,*l1b,*l2W,*l2b,*l3W,*l3b; };

// Padded GEMM driver: packs B + bias, then launches the branch-free kernel.
//   Kp = lda of A (K rounded up to 32); Np = padded N (multiple of 16,
//   chosen >= next consumer's Kp when C feeds another GEMM).
static void gemmP(hipStream_t st,
                  const __hip_bfloat16* A, int Kp, int M,
                  const float* B, int b_nk, int ldb, int N, int K,
                  const float* bias, int Np,
                  float scale, int relu,
                  void* C, int cbf16, int ldc,
                  __hip_bfloat16* BpBuf, float* biasBuf)
{
    int Nt = Np / 16, Kc = Kp / 32;
    int totB = Nt * Kc * 512;
    k_pack_b<<<(totB + 255)/256, 256, 0, st>>>(B, b_nk, ldb, N, K, Kc, totB, BpBuf);
    k_pack_bias<<<1, 256, 0, st>>>(bias, N, Np, biasBuf);
    int Mt = (M + 15) / 16;
    int blocks = (Mt*Nt + 7) / 8;                 // 8 waves / 256 threads per block
    if (cbf16) k_gemm<1><<<blocks,256,0,st>>>(A,Kp,Mt,Nt,Kc,BpBuf,biasBuf,scale,relu,C,ldc);
    else       k_gemm<0><<<blocks,256,0,st>>>(A,Kp,Mt,Nt,Kc,BpBuf,biasBuf,scale,relu,C,ldc);
}

struct Ws {
    __hip_bfloat16 *Dp, *H1, *H2, *H3, *F, *Bp;
    float *biasP;
};

static void ptconv(hipStream_t st,
                   const float* feat, int Cin,
                   const float* P, int Np_,
                   const float* Q, int M, int K,
                   const CvP& cv, float* raw, int* idx, const Ws& ws)
{
    k_knn  <<<(M + 127)/128, 128, 0, st>>>(Q, M, P, Np_, K, idx);
    k_dists<<<(M + 127)/128, 128, 0, st>>>(Q, M, P, idx, K, cv.cen, ws.Dp);
    int R = M * K;   // multiple of 16 at every level
    // neighbor MLP: 48->32->16->16 (bf16 activations, padded lds)
    gemmP(st, ws.Dp, 64, R, cv.l1W, 1, 48, 32, 48, cv.l1b, 32, 1.f, 1, ws.H1, 1, 32, ws.Bp, ws.biasP);
    gemmP(st, ws.H1, 32, R, cv.l2W, 1, 32, 16, 32, cv.l2b, 32, 1.f, 1, ws.H2, 1, 32, ws.Bp, ws.biasP);
    gemmP(st, ws.H2, 32, R, cv.l3W, 1, 16, 16, 16, cv.l3b, 16, 1.f, 1, ws.H3, 1, 16, ws.Bp, ws.biasP);
    // f[m, c, n] outer-product accumulation (bf16, ldF = round32(Cin*16))
    int ldF = (Cin*NC + 31) & ~31;
    long tt = (long)M * Cin;
    k_feinsum<<<(unsigned)((tt + 255)/256), 256, 0, st>>>(feat, Cin, idx, K, ws.H3, M, ws.F, ldF);
    if (ldF > Cin*NC) {
        int span = ldF - Cin*NC;
        k_zero_bf<<<(M*span + 255)/256, 256, 0, st>>>(ws.F, M, ldF, Cin*NC);
    }
    // out = f @ weight / K   (weight is (Cin*16, 96) row-major = KN layout)
    gemmP(st, ws.F, ldF, M, cv.w, 0, C_PL2, C_PL2, Cin*NC, nullptr, C_PL2,
          1.f / (float)K, 0, raw, 0, C_PL2, ws.Bp, ws.biasP);
}

extern "C" void kernel_launch(void* const* d_in, const int* in_sizes, int n_in,
                              void* d_out, int out_size, void* d_ws, size_t ws_size,
                              hipStream_t stream)
{
    (void)in_sizes; (void)n_in; (void)out_size; (void)ws_size;

    auto in = [&](int i) { return (const float*)d_in[i]; };
    // Input flattening (setup_inputs dict insertion order):
    //  0:x  1:input_pts
    //  cv{1,2,3,3d,2d,1d} at 2,10,18,26,34,42 : weight,centers,l1W,l1b,l2W,l2b,l3W,l3b
    //  bn{1,2,3,3d,2d,1d} at 50..61 : g,b
    //  62:fcout.W 63:fcout.b
    //  pnets[h] at 64+18h : c1W,c1b,c2W,c2b,c3W,c3b,f1W,f1b,f2W,f2b,f3W,f3b,
    //                        bn1g,bn1b,bn2g,bn2b,bn3g,bn3b
    auto getcv = [&](int base) {
        CvP c{ in(base), in(base+1), in(base+2), in(base+3),
               in(base+4), in(base+5), in(base+6), in(base+7) };
        return c;
    };
    const float* x_in  = in(0);
    const float* ipts  = in(1);
    CvP cv1 = getcv(2), cv2 = getcv(10), cv3 = getcv(18);
    CvP cv3d = getcv(26), cv2d = getcv(34), cv1d = getcv(42);

    // -------- workspace carve-out (~156 MB worst case) ----------------
    char* w = (char*)d_ws;
    size_t off = 0;
    auto carve = [&](size_t bytes) -> char* {
        char* p = w + off;
        off = (off + bytes + 255) & ~(size_t)255;
        return p;
    };
    int*   sel1 = (int*)carve(64*4);
    int*   sel2 = (int*)carve(16*4);
    int*   sel3 = (int*)carve(8*4);
    float* pts1 = (float*)carve(64*3*4);
    float* pts2 = (float*)carve(16*3*4);
    float* pts3 = (float*)carve(8*3*4);
    int*   idx  = (int*)carve((size_t)N_PTS*8*4);
    Ws ws;
    ws.Dp    = (__hip_bfloat16*)carve((size_t)N_PTS*8*64*2);
    ws.H1    = (__hip_bfloat16*)carve((size_t)N_PTS*8*32*2);
    ws.H2    = (__hip_bfloat16*)carve((size_t)N_PTS*8*32*2);
    ws.H3    = (__hip_bfloat16*)carve((size_t)N_PTS*8*16*2);
    ws.F     = (__hip_bfloat16*)carve((size_t)N_PTS*3072*2);
    ws.Bp    = (__hip_bfloat16*)carve((size_t)6*96*512*2);   // max: cv weight pack
    ws.biasP = (float*)carve(256*4);
    float* raw   = (float*)carve((size_t)N_PTS*C_PL2*4);     // >= 16 rows at every level
    float* x1    = (float*)carve(64*C_PL2*4);
    float* x2    = (float*)carve(16*C_PL2*4);
    float* x3    = (float*)carve(16*C_PL2*4);                // M=8 padded to 16 rows
    float* x3dc  = (float*)carve(16*192*4);
    float* x2dc  = (float*)carve(64*192*4);
    float* x1d   = (float*)carve((size_t)N_PTS*C_PL2*4);
    __hip_bfloat16* x1db = (__hip_bfloat16*)carve((size_t)N_PTS*C_PL2*2);
    float* xoutp = (float*)carve((size_t)N_PTS*16*4);
    int*   cls   = (int*)carve((size_t)N_PTS*4);
    __hip_bfloat16* rf   = (__hip_bfloat16*)carve((size_t)4*32*RF_LD*2);
    float* ph1   = (float*)carve(32*64*4);
    float* ph2   = (float*)carve(32*128*4);
    float* ph3   = (float*)carve(32*256*4);
    __hip_bfloat16* ph1b = (__hip_bfloat16*)carve(32*64*2);
    __hip_bfloat16* ph2b = (__hip_bfloat16*)carve(32*128*2);
    __hip_bfloat16* pg   = (__hip_bfloat16*)carve(16*256*2); // row 0 valid
    __hip_bfloat16* pf1  = (__hip_bfloat16*)carve(16*128*2);
    __hip_bfloat16* pf2  = (__hip_bfloat16*)carve(16*64*2);
    float* headp = (float*)carve(16*16*4);

    // -------- point subset selection ---------------------------------
    k_prep_sel<<<1, 64, 0, stream>>>(ipts, sel1, sel2, sel3, pts1, pts2, pts3);

    // -------- encoder -------------------------------------------------
    ptconv(stream, x_in, 1,   ipts, N_PTS, pts1, 64, 8, cv1, raw, idx, ws);
    k_bn_relu<<<C_PL2, 256, 0, stream>>>(raw, 64, C_PL2, in(50), in(51), x1, C_PL2, nullptr);

    ptconv(stream, x1, C_PL2, pts1, 64,  pts2, 16, 8, cv2, raw, idx, ws);
    k_bn_relu<<<C_PL2, 256, 0, stream>>>(raw, 16, C_PL2, in(52), in(53), x2, C_PL2, nullptr);

    ptconv(stream, x2, C_PL2, pts2, 16,  pts3, 8,  4, cv3, raw, idx, ws);
    k_bn_relu<<<C_PL2, 256, 0, stream>>>(raw, 8, C_PL2, in(54), in(55), x3, C_PL2, nullptr);

    // -------- decoder with skip concats ------------------------------
    ptconv(stream, x3, C_PL2, pts3, 8,   pts2, 16, 4, cv3d, raw, idx, ws);
    k_bn_relu<<<C_PL2, 256, 0, stream>>>(raw, 16, C_PL2, in(56), in(57), x3dc, 192, nullptr);
    k_copy<<<(16*C_PL2 + 255)/256, 256, 0, stream>>>(x2, 16, C_PL2, x3dc + C_PL2, 192);

    ptconv(stream, x3dc, 192, pts2, 16,  pts1, 64, 4, cv2d, raw, idx, ws);
    k_bn_relu<<<C_PL2, 256, 0, stream>>>(raw, 64, C_PL2, in(58), in(59), x2dc, 192, nullptr);
    k_copy<<<(64*C_PL2 + 255)/256, 256, 0, stream>>>(x1, 64, C_PL2, x2dc + C_PL2, 192);

    ptconv(stream, x2dc, 192, pts1, 64,  ipts, N_PTS, 8, cv1d, raw, idx, ws);
    k_bn_relu<<<C_PL2, 256, 0, stream>>>(raw, N_PTS, C_PL2, in(60), in(61), x1d, C_PL2, x1db);

    // -------- classifier head: xout -> d_out[0 .. 16384*5) -----------
    gemmP(stream, x1db, C_PL2, N_PTS, in(62), 1, C_PL2, 5, C_PL2, in(63), 16,
          1.f, 0, xoutp, 0, 16, ws.Bp, ws.biasP);
    float* xout = (float*)d_out;
    k_extract<<<(N_PTS*5 + 255)/256, 256, 0, stream>>>(xoutp, 16, N_PTS, 5, xout, 5);
    k_argmax<<<(N_PTS + 255)/256, 256, 0, stream>>>(xoutp, 16, N_PTS, 5, cls);

    // -------- per-class sampling + PointNet heads --------------------
    k_sample<<<4, 128, 0, stream>>>(cls, N_PTS, x1d, ipts, rf);

    float* heads = (float*)d_out + (size_t)N_PTS*5;
    for (int h = 0; h < 4; ++h) {
        int pb = 64 + 18*h;
        const __hip_bfloat16* rfh = rf + (size_t)h*32*RF_LD;
        gemmP(stream, rfh, RF_LD, 32, in(pb+0), 1, RF_DIM, 64, RF_DIM, in(pb+1), 64,
              1.f, 0, ph1, 0, 64, ws.Bp, ws.biasP);
        k_bn_relu<<<64, 256, 0, stream>>>(ph1, 32, 64, in(pb+12), in(pb+13), ph1, 64, ph1b);
        gemmP(stream, ph1b, 64, 32, in(pb+2), 1, 64, 128, 64, in(pb+3), 128,
              1.f, 0, ph2, 0, 128, ws.Bp, ws.biasP);
        k_bn_relu<<<128, 256, 0, stream>>>(ph2, 32, 128, in(pb+14), in(pb+15), ph2, 128, ph2b);
        gemmP(stream, ph2b, 128, 32, in(pb+4), 1, 128, 256, 128, in(pb+5), 256,
              1.f, 0, ph3, 0, 256, ws.Bp, ws.biasP);
        k_bn_relu<<<256, 256, 0, stream>>>(ph3, 32, 256, in(pb+16), in(pb+17), ph3, 256, nullptr);
        k_meanrows<<<1, 256, 0, stream>>>(ph3, 32, 256, pg);
        gemmP(stream, pg,  256, 1, in(pb+6), 1, 256, 128, 256, in(pb+7), 128,
              1.f, 1, pf1, 1, 128, ws.Bp, ws.biasP);
        gemmP(stream, pf1, 128, 1, in(pb+8), 1, 128, 64, 128, in(pb+9), 64,
              1.f, 1, pf2, 1, 64, ws.Bp, ws.biasP);
        gemmP(stream, pf2, 64, 1, in(pb+10), 1, 64, 4, 64, in(pb+11), 16,
              1.f, 0, headp, 0, 16, ws.Bp, ws.biasP);
        k_extract<<<1, 64, 0, stream>>>(headp, 16, 1, 4, heads + h*4, 4);
    }
}
